// RNNAttentionDecoder_25546465477079
// MI455X (gfx1250) — matrix operand order repaired and checked by reading, not compiled, not run on hardware
//
#include <hip/hip_runtime.h>
#include <hip/hip_bf16.h>
#include <math.h>

// Problem constants (match reference)
#define B_    32
#define T_    64
#define S_    128
#define EMB_  512
#define ENCH_ 1024
#define DECH_ 1024
#define ATT_  512
#define V_    32000
#define VM1_  31999
#define NWG_  32          // persistent decoder workgroups (1 batch row each)

typedef __bf16 v16bf __attribute__((ext_vector_type(16)));
typedef float  v8f   __attribute__((ext_vector_type(8)));

struct alignas(16) U4 { unsigned int x, y, z, w; };
union  FragAB { v16bf v; U4 q[2]; };

__device__ __forceinline__ unsigned short f2bf(float f) {
  unsigned int u = __builtin_bit_cast(unsigned int, f);
  u += 0x7fffu + ((u >> 16) & 1u);           // round-to-nearest-even
  return (unsigned short)(u >> 16);
}

__device__ __forceinline__ v8f zero8() {
  v8f z = {0.f, 0.f, 0.f, 0.f, 0.f, 0.f, 0.f, 0.f};
  return z;
}

// CDNA5 native transcendental tanh (trans pipe). v_nop covers the
// 1-op trans-result hazard (ISA 7.4.3).
__device__ __forceinline__ float fast_tanh(float x) {
  float r;
  asm volatile("v_tanh_f32 %0, %1\n\tv_nop" : "=v"(r) : "v"(x));
  return r;
}

// CDNA5 async global->LDS copy (ASYNCcnt-tracked, no VGPR round trip).
// lds_off = low 32 bits of a generic shared pointer (LDS aperture keeps the
// wave-relative byte offset in addr[31:0], ISA 10.2).
__device__ __forceinline__ void async_ld16(unsigned lds_off,
                                           const unsigned short* gptr) {
  asm volatile("global_load_async_to_lds_b128 %0, %1, off"
               :: "v"(lds_off), "v"(gptr) : "memory");
}
__device__ __forceinline__ void wait_async0() {
  asm volatile("s_wait_asynccnt 0x0" ::: "memory");
}

// A fragment (16x32 bf16, MxK). ISA layout: lane = M row (L&15);
// VGPR0-3 hold K = (L<16 ? 0..7 : 8..15), VGPR4-7 hold K+16.
__device__ __forceinline__ void loadA(FragAB& f, const unsigned short* row0,
                                      int ld, int k0, int lane) {
  const unsigned short* p = row0 + (size_t)(lane & 15) * ld + k0 + ((lane >> 4) << 3);
  f.q[0] = *reinterpret_cast<const U4*>(p);
  f.q[1] = *reinterpret_cast<const U4*>(p + 16);
}
__device__ __forceinline__ void loadA_rowptr(FragAB& f, const unsigned short* rowp,
                                             int k0, int lane) {
  const unsigned short* p = rowp + k0 + ((lane >> 4) << 3);
  f.q[0] = *reinterpret_cast<const U4*>(p);
  f.q[1] = *reinterpret_cast<const U4*>(p + 16);
}
// B fragment (32x16 bf16, KxN) from weight stored [N rows][K cols]:
// lane = N col (L&15); VGPR0-3 hold K = (L<16 ? 0..7 : 8..15), VGPR4-7 hold K+16.
__device__ __forceinline__ void loadB(FragAB& f, const unsigned short* col0,
                                      int ld, int k0, int lane) {
  const unsigned short* p = col0 + (size_t)(lane & 15) * ld + k0 + ((lane >> 4) << 3);
  f.q[0] = *reinterpret_cast<const U4*>(p);
  f.q[1] = *reinterpret_cast<const U4*>(p + 16);
}

__device__ __forceinline__ v8f wmma_bf16(const FragAB& a, const FragAB& b, v8f c) {
  return __builtin_amdgcn_wmma_f32_16x16x32_bf16(
      false, a.v, false, b.v, (short)0, c, false, false);
}

// Device-scope sense-reversal grid barrier for NWG_ co-resident workgroups.
// Agent fences emit gfx1250 cache maintenance (global_wb / global_inv).
__device__ __forceinline__ void grid_barrier(int* cnt, int* gen) {
  __syncthreads();
  if (threadIdx.x == 0) {
    __builtin_amdgcn_fence(__ATOMIC_RELEASE, "agent");
    int g = __hip_atomic_load(gen, __ATOMIC_RELAXED, __HIP_MEMORY_SCOPE_AGENT);
    if (__hip_atomic_fetch_add(cnt, 1, __ATOMIC_RELAXED, __HIP_MEMORY_SCOPE_AGENT) == NWG_ - 1) {
      __hip_atomic_store(cnt, 0, __ATOMIC_RELAXED, __HIP_MEMORY_SCOPE_AGENT);
      __hip_atomic_fetch_add(gen, 1, __ATOMIC_RELEASE, __HIP_MEMORY_SCOPE_AGENT);
    } else {
      while (__hip_atomic_load(gen, __ATOMIC_RELAXED, __HIP_MEMORY_SCOPE_AGENT) == g)
        __builtin_amdgcn_s_sleep(1);
    }
    __builtin_amdgcn_fence(__ATOMIC_ACQUIRE, "agent");
  }
  __syncthreads();
}

// ---------------------------------------------------------------------------
// Kernel 0: zero the barrier words (ws is only poisoned once by the harness)
// ---------------------------------------------------------------------------
__global__ void bar_init_kernel(int* bar) { if (threadIdx.x < 2) bar[threadIdx.x] = 0; }

// ---------------------------------------------------------------------------
// Kernel 1: f32 -> bf16 conversion (grid-stride)
// ---------------------------------------------------------------------------
__global__ __launch_bounds__(256) void cvt_bf16_kernel(const float* __restrict__ src,
                                                       unsigned short* __restrict__ dst,
                                                       long n) {
  long stride = (long)gridDim.x * blockDim.x;
  for (long i = (long)blockIdx.x * blockDim.x + threadIdx.x; i < n; i += stride)
    dst[i] = f2bf(src[i]);
}

// ---------------------------------------------------------------------------
// Kernel 2: enc_proj[B*S, ATT] = enc_bf @ W_enc_bf^T   (hoisted, f32 out)
// ---------------------------------------------------------------------------
__global__ __launch_bounds__(256) void enc_proj_kernel(const unsigned short* __restrict__ encbf,
                                                       const unsigned short* __restrict__ wencbf,
                                                       float* __restrict__ encproj) {
  int lane = threadIdx.x & 31;
  int gw   = (blockIdx.x * blockDim.x + threadIdx.x) >> 5;   // 8192 waves
  int m0 = (gw >> 5) * 16;     // 256 M-tiles (B*S=4096)
  int n0 = (gw & 31) * 16;     // 32  N-tiles (ATT=512)
  v8f c = zero8();
  FragAB a, b;
  for (int k0 = 0; k0 < ENCH_; k0 += 32) {
    loadA(a, encbf  + (size_t)m0 * ENCH_, ENCH_, k0, lane);
    loadB(b, wencbf + (size_t)n0 * ENCH_, ENCH_, k0, lane);
    c = wmma_bf16(a, b, c);
  }
  int n  = n0 + (lane & 15);
  int mb = m0 + ((lane >> 4) << 3);
#pragma unroll
  for (int r = 0; r < 8; ++r)
    encproj[(size_t)(mb + r) * ATT_ + n] = c[r];
}

// ---------------------------------------------------------------------------
// Kernel 3: persistent multi-WGP recurrent decoder.
// 32 workgroups x 256 threads. WG g owns batch row b=g for the whole
// attention block (scores -> softmax -> context: WG-local, no global sync).
// 4 grid barriers per step cover the 4 true cross-WG dependency edges.
// h0/h1 state is ping-ponged (bf16, in L2) to avoid WAR barriers.
// ---------------------------------------------------------------------------
__global__ __launch_bounds__(256) void decoder_kernel(
    const int* __restrict__ y,                  // [B,T]
    const float* __restrict__ enc,              // [B,S,ENCH] f32
    const float* __restrict__ h_init,           // [2,B,DECH] f32
    const int* __restrict__ mask,               // [B,S]
    const unsigned short* __restrict__ embbf,   // [V,EMB] bf16
    const unsigned short* __restrict__ wdecbf,  // [ATT,DECH]
    const float* __restrict__ v_w,              // [ATT]
    const unsigned short* __restrict__ wih0bf,  // [DECH, EMB+ENCH]
    const unsigned short* __restrict__ whh0bf,  // [DECH, DECH]
    const float* __restrict__ bih0, const float* __restrict__ bhh0,
    const unsigned short* __restrict__ wih1bf,  // [DECH, DECH]
    const unsigned short* __restrict__ whh1bf,  // [DECH, DECH]
    const float* __restrict__ bih1, const float* __restrict__ bhh1,
    const float* __restrict__ encproj,          // [B*S, ATT] f32
    unsigned short* __restrict__ h0g,           // [2][B,DECH] bf16 ping-pong
    unsigned short* __restrict__ h1g,           // [2][B,DECH] bf16 ping-pong
    unsigned short* __restrict__ ctxg,          // [B,ENCH] bf16
    float* __restrict__ decpg,                  // [B,ATT] f32
    int* __restrict__ bar,                      // {cnt, gen}
    unsigned short* __restrict__ Hbf,           // [B*T, DECH] bf16 out
    float* __restrict__ attn_out)               // [B,T,S] f32 out
{
  __shared__ float ls_part[256];
  __shared__ float ls_sc[S_];

  const int tid = threadIdx.x, lane = tid & 31, wave = tid >> 5;
  const int g = blockIdx.x;                 // this WG's batch row
  int* cnt = bar; int* gen = bar + 1;

  // init recurrent state (f32 -> bf16) across the grid
  for (int i = blockIdx.x * 256 + tid; i < B_ * DECH_; i += NWG_ * 256) {
    h0g[i] = f2bf(h_init[i]);
    h1g[i] = f2bf(h_init[B_ * DECH_ + i]);
  }
  grid_barrier(cnt, gen);

  for (int t = 0; t < T_; ++t) {
    const int cur = t & 1, nxt = cur ^ 1;
    const unsigned short* h0cur = h0g + cur * B_ * DECH_;
    unsigned short*       h0nxt = h0g + nxt * B_ * DECH_;
    const unsigned short* h1cur = h1g + cur * B_ * DECH_;
    unsigned short*       h1nxt = h1g + nxt * B_ * DECH_;

    // ---- P1: dec_p[32,512] = h1 @ W_dec^T  (64 tiles; 2 per WG) ----
    if (wave < 2) {
      int tile = g * 2 + wave;
      int m0 = (tile >> 5) * 16, n0 = (tile & 31) * 16;
      v8f c = zero8();
      FragAB a, b;
      for (int k0 = 0; k0 < DECH_; k0 += 32) {
        loadA(a, h1cur + (size_t)m0 * DECH_, DECH_, k0, lane);
        loadB(b, wdecbf + (size_t)n0 * DECH_, DECH_, k0, lane);
        c = wmma_bf16(a, b, c);
      }
      int n = n0 + (lane & 15), mb = m0 + ((lane >> 4) << 3);
#pragma unroll
      for (int r = 0; r < 8; ++r) decpg[(mb + r) * ATT_ + n] = c[r];
    }
    grid_barrier(cnt, gen);

    // ---- P2: scores[g,s] = v . tanh(enc_proj[g,s,:] + dec_p[g,:]) ----
    // 128 pairs, 2 threads per pair (a-dim halves), float4 streams.
    {
      int pr = tid >> 1, half = tid & 1;
      const float4* ep = (const float4*)(encproj + ((size_t)g * S_ + pr) * ATT_ + half * 256);
      const float4* dp = (const float4*)(decpg + (size_t)g * ATT_ + half * 256);
      const float4* vp = (const float4*)(v_w + half * 256);
      float acc = 0.f;
      for (int i = 0; i < 64; ++i) {
        float4 e = ep[i], d = dp[i], v = vp[i];
        acc += v.x * fast_tanh(e.x + d.x) + v.y * fast_tanh(e.y + d.y)
             + v.z * fast_tanh(e.z + d.z) + v.w * fast_tanh(e.w + d.w);
      }
      ls_part[tid] = acc;
      __syncthreads();
      if (half == 0) {
        float sc = ls_part[tid] + ls_part[tid + 1];
        ls_sc[pr] = (mask[g * S_ + pr] == 0) ? -1e9f : sc;
      }
      __syncthreads();
    }

    // ---- P3: softmax over S=128 (wave 0), write attn weights ----
    if (wave == 0) {
      float vals[4];
      float mx = -INFINITY;
#pragma unroll
      for (int i = 0; i < 4; ++i) { vals[i] = ls_sc[lane + 32 * i]; mx = fmaxf(mx, vals[i]); }
      for (int o = 16; o > 0; o >>= 1) mx = fmaxf(mx, __shfl_xor(mx, o, 32));
      float sum = 0.f;
#pragma unroll
      for (int i = 0; i < 4; ++i) { vals[i] = __expf(vals[i] - mx); sum += vals[i]; }
      for (int o = 16; o > 0; o >>= 1) sum += __shfl_xor(sum, o, 32);
      float inv = 1.f / sum;
#pragma unroll
      for (int i = 0; i < 4; ++i) {
        float w = vals[i] * inv;
        ls_sc[lane + 32 * i] = w;
        attn_out[((size_t)g * T_ + t) * S_ + lane + 32 * i] = w;
      }
    }
    __syncthreads();

    // ---- P4: context[g,e] = sum_s attn[s]*enc[g,s,e]  (bf16 to L2) ----
    {
      int e0 = tid * 4;
      const float* eb = enc + (size_t)g * S_ * ENCH_ + e0;
      float a0 = 0.f, a1 = 0.f, a2 = 0.f, a3 = 0.f;
      for (int s = 0; s < S_; ++s) {
        float w = ls_sc[s];
        float4 e = *(const float4*)(eb + (size_t)s * ENCH_);
        a0 += w * e.x; a1 += w * e.y; a2 += w * e.z; a3 += w * e.w;
      }
      unsigned short* cp = ctxg + (size_t)g * ENCH_ + e0;
      cp[0] = f2bf(a0); cp[1] = f2bf(a1); cp[2] = f2bf(a2); cp[3] = f2bf(a3);
    }
    grid_barrier(cnt, gen);

    // ---- P5: h0 = tanh([emb|ctx] @ W_ih0^T + h0 @ W_hh0^T + b)  (128 tiles) ----
    if (wave < 4) {
      int tile = g * 4 + wave;
      int m0 = (tile >> 6) * 16, n0 = (tile & 63) * 16;
      v8f c = zero8();
      FragAB a, b;
      const unsigned short* erow = embbf + (size_t)y[(m0 + (lane & 15)) * T_ + t] * EMB_;
      const unsigned short* wx = wih0bf + (size_t)n0 * (EMB_ + ENCH_);
      for (int k0 = 0; k0 < EMB_; k0 += 32) {            // embedding slice (gather)
        loadA_rowptr(a, erow, k0, lane);
        loadB(b, wx, EMB_ + ENCH_, k0, lane);
        c = wmma_bf16(a, b, c);
      }
      for (int k0 = 0; k0 < ENCH_; k0 += 32) {           // context slice
        loadA(a, ctxg + (size_t)m0 * ENCH_, ENCH_, k0, lane);
        loadB(b, wx, EMB_ + ENCH_, EMB_ + k0, lane);
        c = wmma_bf16(a, b, c);
      }
      const unsigned short* wh = whh0bf + (size_t)n0 * DECH_;
      for (int k0 = 0; k0 < DECH_; k0 += 32) {           // recurrent slice
        loadA(a, h0cur + (size_t)m0 * DECH_, DECH_, k0, lane);
        loadB(b, wh, DECH_, k0, lane);
        c = wmma_bf16(a, b, c);
      }
      int n = n0 + (lane & 15), mb = m0 + ((lane >> 4) << 3);
      float bias = bih0[n] + bhh0[n];
#pragma unroll
      for (int r = 0; r < 8; ++r)
        h0nxt[(mb + r) * DECH_ + n] = f2bf(fast_tanh(c[r] + bias));
    }
    grid_barrier(cnt, gen);

    // ---- P6: h1 = tanh(h0 @ W_ih1^T + h1 @ W_hh1^T + b)  (128 tiles) ----
    if (wave < 4) {
      int tile = g * 4 + wave;
      int m0 = (tile >> 6) * 16, n0 = (tile & 63) * 16;
      v8f c = zero8();
      FragAB a, b;
      const unsigned short* wi = wih1bf + (size_t)n0 * DECH_;
      const unsigned short* wh = whh1bf + (size_t)n0 * DECH_;
      for (int k0 = 0; k0 < DECH_; k0 += 32) {
        loadA(a, h0nxt + (size_t)m0 * DECH_, DECH_, k0, lane);
        loadB(b, wi, DECH_, k0, lane);
        c = wmma_bf16(a, b, c);
      }
      for (int k0 = 0; k0 < DECH_; k0 += 32) {
        loadA(a, h1cur + (size_t)m0 * DECH_, DECH_, k0, lane);
        loadB(b, wh, DECH_, k0, lane);
        c = wmma_bf16(a, b, c);
      }
      int n = n0 + (lane & 15), mb = m0 + ((lane >> 4) << 3);
      float bias = bih1[n] + bhh1[n];
#pragma unroll
      for (int r = 0; r < 8; ++r) {
        unsigned short hv = f2bf(fast_tanh(c[r] + bias));
        int bb = mb + r;
        h1nxt[bb * DECH_ + n] = hv;
        Hbf[((size_t)bb * T_ + t) * DECH_ + n] = hv;   // row b*T+t for fc GEMM
      }
    }
    grid_barrier(cnt, gen);
  }
}

// ---------------------------------------------------------------------------
// Kernel 4: out[B*T, VM1] = H_bf @ fc_w_bf^T + fc_b   (134 GFLOP)
// Block = 8 waves = 8 M-tiles (128 rows) x 128 cols. B is staged through a
// double-buffered LDS (2x8KB) via CDNA5 GLOBAL_LOAD_ASYNC_TO_LDS_B128:
// the next K-chunk streams into LDS while the current chunk feeds 8 WMMAs,
// with s_wait_asynccnt 0 + one barrier per K-step.
// ---------------------------------------------------------------------------
__global__ __launch_bounds__(256) void fc_kernel(const unsigned short* __restrict__ Hbf,
                                                 const unsigned short* __restrict__ fcwbf,
                                                 const float* __restrict__ fcb,
                                                 float* __restrict__ out) {
  __shared__ unsigned short ldsB[2][128 * 32];   // double-buffered 8 KB stages
  int tid = threadIdx.x, lane = tid & 31, wave = tid >> 5;
  int mg = blockIdx.x / 250;                  // 16 groups of 8 M-tiles (B*T=2048)
  int ng = blockIdx.x % 250;                  // 250 groups of 128 columns
  int m0 = (mg * 8 + wave) * 16;
  int n0base = ng * 128;

  v8f c[8] = {zero8(), zero8(), zero8(), zero8(), zero8(), zero8(), zero8(), zero8()};
  const unsigned short* arow = Hbf + (size_t)m0 * DECH_;

  // staging map: thread -> (row n, 32B of K), two async b128 per thread
  int sn = tid >> 1, sk = (tid & 1) << 4;
  const unsigned short* bsrc = fcwbf + (size_t)(n0base + sn) * DECH_ + sk;
  unsigned lds0 = (unsigned)(uintptr_t)(&ldsB[0][sn * 32 + sk]);
  unsigned lds1 = (unsigned)(uintptr_t)(&ldsB[1][sn * 32 + sk]);

  // prologue: stage K-chunk 0 into buffer 0
  async_ld16(lds0,      bsrc);
  async_ld16(lds0 + 16, bsrc + 8);
  wait_async0();
  __syncthreads();

  int buf = 0;
  for (int k0 = 0; k0 < DECH_; k0 += 32) {
    // stage next chunk into the other buffer; overlaps with WMMA below
    if (k0 + 32 < DECH_) {
      unsigned dst = buf ? lds0 : lds1;
      async_ld16(dst,      bsrc + k0 + 32);
      async_ld16(dst + 16, bsrc + k0 + 40);
    }
    FragAB a, b;
    loadA(a, arow, DECH_, k0, lane);
#pragma unroll
    for (int j = 0; j < 8; ++j) {
      loadB(b, &ldsB[buf][j * 16 * 32], 32, 0, lane);
      c[j] = wmma_bf16(a, b, c[j]);
    }
    wait_async0();        // next stage landed (issued before the compute)
    __syncthreads();      // all waves done with ldsB[buf]; stage visible
    buf ^= 1;
  }

  int mb = m0 + ((lane >> 4) << 3);
#pragma unroll
  for (int j = 0; j < 8; ++j) {
    int n = n0base + j * 16 + (lane & 15);
    if (n < VM1_) {
      float bias = fcb[n];
#pragma unroll
      for (int r = 0; r < 8; ++r)
        out[(size_t)(mb + r) * VM1_ + n] = c[j][r] + bias;
    }
  }
}

// ---------------------------------------------------------------------------
extern "C" void kernel_launch(void* const* d_in, const int* in_sizes, int n_in,
                              void* d_out, int out_size, void* d_ws, size_t ws_size,
                              hipStream_t stream) {
  const int*   y      = (const int*)  d_in[0];
  const float* enc    = (const float*)d_in[1];
  const float* h_init = (const float*)d_in[2];
  const int*   mask   = (const int*)  d_in[3];
  const float* emb    = (const float*)d_in[4];
  const float* W_enc  = (const float*)d_in[5];
  const float* W_dec  = (const float*)d_in[6];
  const float* v_w    = (const float*)d_in[7];
  const float* W_ih0  = (const float*)d_in[8];
  const float* W_hh0  = (const float*)d_in[9];
  const float* b_ih0  = (const float*)d_in[10];
  const float* b_hh0  = (const float*)d_in[11];
  const float* W_ih1  = (const float*)d_in[12];
  const float* W_hh1  = (const float*)d_in[13];
  const float* b_ih1  = (const float*)d_in[14];
  const float* b_hh1  = (const float*)d_in[15];
  const float* fc_w   = (const float*)d_in[16];
  const float* fc_b   = (const float*)d_in[17];

  float* outputs  = (float*)d_out;                        // [B,T,V-1]
  float* attn_out = outputs + (size_t)B_ * T_ * VM1_;     // [B,T,S]

  // Workspace partition (256B aligned)
  char* ws = (char*)d_ws;
  size_t off = 0;
  auto take = [&](size_t bytes) { char* p = ws + off; off += (bytes + 255) & ~(size_t)255; return p; };
  unsigned short* emb_bf  = (unsigned short*)take((size_t)V_ * EMB_ * 2);
  unsigned short* enc_bf  = (unsigned short*)take((size_t)B_ * S_ * ENCH_ * 2);
  unsigned short* wenc_bf = (unsigned short*)take((size_t)ATT_ * ENCH_ * 2);
  unsigned short* wdec_bf = (unsigned short*)take((size_t)ATT_ * DECH_ * 2);
  unsigned short* wih0_bf = (unsigned short*)take((size_t)DECH_ * (EMB_ + ENCH_) * 2);
  unsigned short* whh0_bf = (unsigned short*)take((size_t)DECH_ * DECH_ * 2);
  unsigned short* wih1_bf = (unsigned short*)take((size_t)DECH_ * DECH_ * 2);
  unsigned short* whh1_bf = (unsigned short*)take((size_t)DECH_ * DECH_ * 2);
  unsigned short* fcw_bf  = (unsigned short*)take((size_t)V_ * DECH_ * 2);   // 32000 rows reserved
  float*          encproj = (float*)take((size_t)B_ * S_ * ATT_ * 4);
  unsigned short* Hbf     = (unsigned short*)take((size_t)B_ * T_ * DECH_ * 2);
  unsigned short* h0g     = (unsigned short*)take((size_t)2 * B_ * DECH_ * 2);
  unsigned short* h1g     = (unsigned short*)take((size_t)2 * B_ * DECH_ * 2);
  unsigned short* ctxg    = (unsigned short*)take((size_t)B_ * ENCH_ * 2);
  float*          decpg   = (float*)take((size_t)B_ * ATT_ * 4);
  int*            bar     = (int*)take(256);

  // 0) barrier init (ws poison is not re-applied between replays)
  bar_init_kernel<<<1, 32, 0, stream>>>(bar);

  // 1) bf16 conversions
  auto cvt = [&](const float* s, unsigned short* d, long n) {
    cvt_bf16_kernel<<<4096, 256, 0, stream>>>(s, d, n);
  };
  cvt(emb,   emb_bf,  (long)V_ * EMB_);
  cvt(enc,   enc_bf,  (long)B_ * S_ * ENCH_);
  cvt(W_enc, wenc_bf, (long)ATT_ * ENCH_);
  cvt(W_dec, wdec_bf, (long)ATT_ * DECH_);
  cvt(W_ih0, wih0_bf, (long)DECH_ * (EMB_ + ENCH_));
  cvt(W_hh0, whh0_bf, (long)DECH_ * DECH_);
  cvt(W_ih1, wih1_bf, (long)DECH_ * DECH_);
  cvt(W_hh1, whh1_bf, (long)DECH_ * DECH_);
  cvt(fc_w,  fcw_bf,  (long)VM1_ * DECH_);

  // 2) hoisted encoder projection: 8192 tiles, 8 waves/block
  enc_proj_kernel<<<1024, 256, 0, stream>>>(enc_bf, wenc_bf, encproj);

  // 3) persistent recurrence: 32 WGs, 4 grid barriers per step
  decoder_kernel<<<NWG_, 256, 0, stream>>>(
      y, enc, h_init, mask, emb_bf, wdec_bf, v_w,
      wih0_bf, whh0_bf, b_ih0, b_hh0,
      wih1_bf, whh1_bf, b_ih1, b_hh1,
      encproj, h0g, h1g, ctxg, decpg, bar, Hbf, attn_out);

  // 4) deferred vocabulary GEMM: 4000 blocks, async double-buffered B staging
  fc_kernel<<<4000, 256, 0, stream>>>(Hbf, fcw_bf, fc_b, outputs);
}
